// StateSpaceLayer_40870908789225
// MI455X (gfx1250) — matrix operand order repaired and checked by reading, not compiled, not run on hardware
//
#include <hip/hip_runtime.h>
#include <hip/hip_bf16.h>

typedef __attribute__((ext_vector_type(16))) __bf16 v16bf;
typedef __attribute__((ext_vector_type(8)))  __bf16 v8bf;
typedef __attribute__((ext_vector_type(8)))  float  v8f;
typedef __attribute__((ext_vector_type(4)))  int    v4i;

#define DM 1024
#define DS 64
#define NB 4
#define TSEQ 4096
#define NTOK (NB * TSEQ)      // 16384 tokens
#define CHUNK 64
#define NCHUNK (TSEQ / CHUNK) // 64 chunks per batch

#if __has_builtin(__builtin_amdgcn_global_load_async_to_lds_b128)
#define HAVE_ASYNC_LDS 1
#else
#define HAVE_ASYNC_LDS 0
#endif

#if __has_builtin(__builtin_amdgcn_s_wait_asynccnt)
#define WAIT_ASYNC() __builtin_amdgcn_s_wait_asynccnt(0)
#else
#define WAIT_ASYNC() asm volatile("s_wait_asynccnt 0" ::: "memory")
#endif

typedef __attribute__((address_space(1))) v4i as1_v4i;   // global int4*
typedef __attribute__((address_space(3))) v4i as3_v4i;   // LDS int4*

// generic->AS1: global addresses are bit-identical in generic and AS1
__device__ __forceinline__ as1_v4i* to_glb(const void* p) {
    return (as1_v4i*)(unsigned long long)p;
}
// generic->AS3: generic LDS addr = {aperture_hi32, lds_offset_lo32}
__device__ __forceinline__ as3_v4i* to_lds(const void* p) {
    return (as3_v4i*)(unsigned int)(unsigned long long)p;
}

__device__ __forceinline__ unsigned short f2bf(float f) {
    unsigned int u = __float_as_uint(f);
    u += 0x7FFFu + ((u >> 16) & 1u);   // round to nearest even
    return (unsigned short)(u >> 16);
}

union AFrag { v16bf v; v8bf h[2]; };

// ---------------- conversion kernels ----------------

// x f32 -> xb bf16 (same row-major layout), 4 elements/thread
__global__ void k_cvt_x(const float* __restrict__ x, unsigned short* __restrict__ xb) {
    const int i = (blockIdx.x * 256 + threadIdx.x) * 4;
    const float4 f = *(const float4*)(x + i);
    ushort4 o;
    o.x = f2bf(f.x); o.y = f2bf(f.y); o.z = f2bf(f.z); o.w = f2bf(f.w);
    *(ushort4*)(xb + i) = o;
}

// in[K][N] f32  ->  out[N][K] bf16  (transpose so B-fragments load contiguously)
__global__ void k_cvt_t(const float* __restrict__ in, unsigned short* __restrict__ out,
                        int K, int N) {
    const int i = blockIdx.x * 256 + threadIdx.x;
    if (i >= K * N) return;
    const int n = i / K, k = i % K;
    out[i] = f2bf(in[k * N + n]);
}

// ---------------- u = x @ B  (bf16 WMMA, f32 out) ----------------
// one wave per 16-token strip -> 16x64; register ping-pong pipelined over K
__global__ __launch_bounds__(256) void k_gemm_u(const unsigned short* __restrict__ xb,
                                                const unsigned short* __restrict__ Bt,
                                                float* __restrict__ u) {
    const int tid  = threadIdx.x;
    const int wave = (blockIdx.x * 256 + tid) >> 5;   // 0..1023
    const int l    = tid & 31;
    const int lm   = l & 15;
    const int khi  = (l & 16) ? 8 : 0;
    const int kb   = (l & 16) ? 16 : 0;
    v8f acc[4] = {};
    const unsigned short* aRow = xb + (wave * 16 + lm) * DM;

    auto loadA = [&](int ks, AFrag& a) {
        const int k0 = ks * 32;
        a.h[0] = *(const v8bf*)(aRow + k0 + khi);
        a.h[1] = *(const v8bf*)(aRow + k0 + 16 + khi);
    };
    auto loadB = [&](int ks, AFrag* b) {
        const int k0 = ks * 32;
#pragma unroll
        for (int nt = 0; nt < 4; ++nt) {
            const unsigned short* bRow = Bt + (nt * 16 + lm) * DM + k0 + khi;
            b[nt].h[0] = *(const v8bf*)(bRow);
            b[nt].h[1] = *(const v8bf*)(bRow + 16);
        }
    };

    // ping-pong register buffers: with unroll-2 the buffer index is a
    // compile-time constant in each copy -> no register moves
    AFrag aF[2]; AFrag bF[2][4];
    loadA(0, aF[0]); loadB(0, bF[0]);
#pragma unroll 2
    for (int ks = 0; ks < DM / 32; ++ks) {
        const int cur = ks & 1, nxt = cur ^ 1;
        if (ks < DM / 32 - 1) { loadA(ks + 1, aF[nxt]); loadB(ks + 1, bF[nxt]); }
#pragma unroll
        for (int nt = 0; nt < 4; ++nt)
            acc[nt] = __builtin_amdgcn_wmma_f32_16x16x32_bf16(
                false, aF[cur].v, false, bF[cur][nt].v, (short)0, acc[nt], false, false);
    }
    const int mrow = wave * 16 + ((l & 16) ? 8 : 0);
#pragma unroll
    for (int nt = 0; nt < 4; ++nt)
#pragma unroll
        for (int r = 0; r < 8; ++r)
            u[(mrow + r) * DS + nt * 16 + lm] = acc[nt][r];
}

// ---------------- chunked linear scan h_t = a*h_{t-1} + u_t ----------------

__global__ void k_scan1(const float* __restrict__ u, const float* __restrict__ Ad,
                        float* __restrict__ carry) {
    const int s = threadIdx.x;                 // 0..63 (state, coalesced)
    const int bc = blockIdx.x;                 // b*NCHUNK + chunk
    const int b = bc / NCHUNK, chunk = bc % NCHUNK;
    const float a = 1.f / (1.f + __expf(-Ad[s]));
    const float* up = u + ((b * TSEQ + chunk * CHUNK) * DS + s);
    float h = 0.f;
    for (int i = 0; i < CHUNK; ++i) h = fmaf(a, h, up[i * DS]);
    carry[bc * DS + s] = h;
}

__global__ void k_scan2(const float* __restrict__ carry, const float* __restrict__ Ad,
                        float* __restrict__ hin) {
    const int tid = threadIdx.x;               // 256 = 4 batches * 64 states
    const int b = tid >> 6, s = tid & 63;
    const float a = 1.f / (1.f + __expf(-Ad[s]));
    float aL = a;
#pragma unroll
    for (int i = 0; i < 6; ++i) aL *= aL;      // a^64
    float h = 0.f;
    hin[(b * NCHUNK) * DS + s] = 0.f;
    for (int c = 1; c < NCHUNK; ++c) {
        h = fmaf(aL, h, carry[(b * NCHUNK + c - 1) * DS + s]);
        hin[(b * NCHUNK + c) * DS + s] = h;
    }
}

__global__ void k_scan3(const float* __restrict__ u, const float* __restrict__ Ad,
                        const float* __restrict__ hin, unsigned short* __restrict__ hb) {
    const int s = threadIdx.x;
    const int bc = blockIdx.x;
    const int b = bc / NCHUNK, chunk = bc % NCHUNK;
    const float a = 1.f / (1.f + __expf(-Ad[s]));
    const int base = (b * TSEQ + chunk * CHUNK) * DS + s;
    float h = hin[bc * DS + s];
    for (int i = 0; i < CHUNK; ++i) {
        h = fmaf(a, h, u[base + i * DS]);
        hb[base + i * DS] = f2bf(h);
    }
}

// ---------------- y = xb @ Dt + hb @ Ct  (fused K=1024+64 GEMM) ----------------
// 256 threads = 8 waves, 128x128 tile; wave = 64x32.
// Double-buffered LDS; tiles staged with GLOBAL_LOAD_ASYNC_TO_LDS_B128 (ASYNCcnt)
// so k+1 tile loads overlap the WMMA chain on tile k.
__global__ __launch_bounds__(256) void k_gemm_y(
    const unsigned short* __restrict__ xb, const unsigned short* __restrict__ Dt,
    const unsigned short* __restrict__ hb, const unsigned short* __restrict__ Ct,
    float* __restrict__ y) {
    __shared__ __align__(16) unsigned short As[2][128 * 32];
    __shared__ __align__(16) unsigned short Bs[2][128 * 32];
    const int tid = threadIdx.x;
    const int tokBase = blockIdx.x * 128;
    const int nBase   = blockIdx.y * 128;
    const int w = tid >> 5, l = tid & 31;
    const int wm = w >> 2, wn = w & 3;
    const int lm = l & 15;
    const int khi = (l & 16) ? 8 : 0;
    const int kb  = (l & 16) ? 16 : 0;
    v8f acc[4][2] = {};

    auto stage = [&](int ks, int buf) {
        const unsigned short* sa; const unsigned short* sb;
        int ldA, ldB, kOff;
        if (ks < 32) { sa = xb; ldA = DM; sb = Dt; ldB = DM; kOff = ks * 32; }
        else         { sa = hb; ldA = DS; sb = Ct; ldB = DS; kOff = (ks - 32) * 32; }
#pragma unroll
        for (int i = 0; i < 2; ++i) {
            const int c = tid + i * 256;           // 0..511
            const int r = c >> 2, off = (c & 3) * 8;
#if HAVE_ASYNC_LDS
            __builtin_amdgcn_global_load_async_to_lds_b128(
                to_glb(&sa[(tokBase + r) * ldA + kOff + off]),
                to_lds(&As[buf][r * 32 + off]), 0, 0);
            __builtin_amdgcn_global_load_async_to_lds_b128(
                to_glb(&sb[(nBase + r) * ldB + kOff + off]),
                to_lds(&Bs[buf][r * 32 + off]), 0, 0);
#else
            *(uint4*)&As[buf][r * 32 + off] = *(const uint4*)&sa[(tokBase + r) * ldA + kOff + off];
            *(uint4*)&Bs[buf][r * 32 + off] = *(const uint4*)&sb[(nBase + r) * ldB + kOff + off];
#endif
        }
    };

    stage(0, 0);
#if HAVE_ASYNC_LDS
    WAIT_ASYNC();
#endif
    __syncthreads();

    for (int ks = 0; ks < 34; ++ks) {
        const int cur = ks & 1;
        if (ks + 1 < 34) stage(ks + 1, cur ^ 1);   // overlap with compute below

        AFrag a[4], b[2];
#pragma unroll
        for (int mt = 0; mt < 4; ++mt) {
            const int rm = (wm * 64 + mt * 16 + lm) * 32;
            a[mt].h[0] = *(const v8bf*)&As[cur][rm + khi];
            a[mt].h[1] = *(const v8bf*)&As[cur][rm + 16 + khi];
        }
#pragma unroll
        for (int nt = 0; nt < 2; ++nt) {
            const int rn = (wn * 32 + nt * 16 + lm) * 32;
            b[nt].h[0] = *(const v8bf*)&Bs[cur][rn + khi];
            b[nt].h[1] = *(const v8bf*)&Bs[cur][rn + 16 + khi];
        }
#pragma unroll
        for (int mt = 0; mt < 4; ++mt)
#pragma unroll
            for (int nt = 0; nt < 2; ++nt)
                acc[mt][nt] = __builtin_amdgcn_wmma_f32_16x16x32_bf16(
                    false, a[mt].v, false, b[nt].v, (short)0, acc[mt][nt], false, false);

#if HAVE_ASYNC_LDS
        WAIT_ASYNC();                              // own async loads for buf cur^1 done
#endif
        __syncthreads();                           // all waves done reading/writing
    }

    const int mhalf = (l & 16) ? 8 : 0;
#pragma unroll
    for (int mt = 0; mt < 4; ++mt)
#pragma unroll
        for (int nt = 0; nt < 2; ++nt)
#pragma unroll
            for (int r = 0; r < 8; ++r) {
                const int row = tokBase + wm * 64 + mt * 16 + mhalf + r;
                const int col = nBase + wn * 32 + nt * 16 + lm;
                y[row * DM + col] = acc[mt][nt][r];
            }
}

// ---------------- host launch ----------------

extern "C" void kernel_launch(void* const* d_in, const int* in_sizes, int n_in,
                              void* d_out, int out_size, void* d_ws, size_t ws_size,
                              hipStream_t stream) {
    (void)in_sizes; (void)n_in; (void)out_size; (void)ws_size;
    const float* x  = (const float*)d_in[0];
    const float* Ad = (const float*)d_in[1];
    const float* Bp = (const float*)d_in[2];
    const float* Cp = (const float*)d_in[3];
    const float* Dp = (const float*)d_in[4];
    float* y = (float*)d_out;

    char* ws = (char*)d_ws;
    size_t off = 0;
    auto alloc = [&](size_t bytes) {
        void* p = ws + off;
        off = (off + bytes + 255) & ~size_t(255);
        return p;
    };
    unsigned short* xb = (unsigned short*)alloc((size_t)NTOK * DM * 2); // bf16 x
    unsigned short* Dt = (unsigned short*)alloc((size_t)DM * DM * 2);   // bf16 D^T [n][k]
    unsigned short* Ct = (unsigned short*)alloc((size_t)DM * DS * 2);   // bf16 C^T [n][k]
    unsigned short* Bt = (unsigned short*)alloc((size_t)DS * DM * 2);   // bf16 B^T [s][k]
    unsigned short* hb = (unsigned short*)alloc((size_t)NTOK * DS * 2); // bf16 h
    float* u     = (float*)alloc((size_t)NTOK * DS * 4);
    float* carry = (float*)alloc((size_t)NB * NCHUNK * DS * 4);
    float* hin   = (float*)alloc((size_t)NB * NCHUNK * DS * 4);

    k_cvt_x<<<(NTOK * DM) / 1024, 256, 0, stream>>>(x, xb);
    k_cvt_t<<<(DM * DM + 255) / 256, 256, 0, stream>>>(Dp, Dt, DM, DM); // D: K=1024,N=1024
    k_cvt_t<<<(DS * DM + 255) / 256, 256, 0, stream>>>(Cp, Ct, DS, DM); // C: K=64,  N=1024
    k_cvt_t<<<(DM * DS + 255) / 256, 256, 0, stream>>>(Bp, Bt, DM, DS); // B: K=1024,N=64

    k_gemm_u<<<128, 256, 0, stream>>>(xb, Bt, u);

    k_scan1<<<NB * NCHUNK, CHUNK, 0, stream>>>(u, Ad, carry);
    k_scan2<<<1, 256, 0, stream>>>(carry, Ad, hin);
    k_scan3<<<NB * NCHUNK, CHUNK, 0, stream>>>(u, Ad, hin, hb);

    dim3 grid(NTOK / 128, DM / 128);
    k_gemm_y<<<grid, 256, 0, stream>>>(xb, Dt, hb, Ct, y);
}
